// DifferentiableLengthRegulator_50190987821755
// MI455X (gfx1250) — matrix operand order repaired and checked by reading, not compiled, not run on hardware
//
#include <hip/hip_runtime.h>

typedef __attribute__((ext_vector_type(16))) _Float16 v16h;
typedef __attribute__((ext_vector_type(8)))  _Float16 v8h;
typedef __attribute__((ext_vector_type(8)))  float    v8f;

#define B_ 16
#define L_ 512
#define D_ 384
#define INV_TEMP 10.0f

// ---------------- Kernel 1: per-row inclusive cumsum of durations ----------------
__global__ void lr_cumsum(const float* __restrict__ dur, float* __restrict__ cum) {
  __shared__ float s[L_];
  int b = blockIdx.x, t = threadIdx.x;
  s[t] = dur[b * L_ + t];
  __syncthreads();
  for (int off = 1; off < L_; off <<= 1) {
    float add = (t >= off) ? s[t - off] : 0.0f;
    __syncthreads();
    s[t] += add;
    __syncthreads();
  }
  cum[b * L_ + t] = s[t];
}

// ------- Kernel 2: features [B,L,D] f32 -> Ft [B,D,L] f16 (LDS tile transpose) ----
__global__ void lr_transpose(const float* __restrict__ feat, _Float16* __restrict__ ft) {
  __shared__ float tile[32][33];
  int b = blockIdx.z;
  int l0 = blockIdx.x * 32, d0 = blockIdx.y * 32;
  int tx = threadIdx.x, ty = threadIdx.y;  // blockDim = (32, 8)
  for (int j = 0; j < 32; j += 8)
    tile[ty + j][tx] = feat[((size_t)b * L_ + (l0 + ty + j)) * D_ + (d0 + tx)];
  __syncthreads();
  for (int j = 0; j < 32; j += 8)
    ft[(size_t)b * D_ * L_ + (size_t)(d0 + ty + j) * L_ + (l0 + tx)] =
        (_Float16)tile[tx][ty + j];
}

// --- Kernel 3: one wave per (b,t): softmax over L, write normalized f16 weights ---
__global__ void lr_weights(const float* __restrict__ cum, const float* __restrict__ dur,
                           _Float16* __restrict__ W, int T, int Tpad) {
  int lane = threadIdx.x & 31;
  int wave = threadIdx.x >> 5;
  int t = blockIdx.x * 8 + wave;
  int b = blockIdx.y;
  if (t >= T) return;

  float tc = (float)t + 0.5f;
  float dist[16];
  bool  valid[16];
  float dmin = 3.0e38f;
#pragma unroll
  for (int j = 0; j < 16; ++j) {
    int l = lane + 32 * j;                 // coalesced, stride-32 over L
    float c = cum[b * L_ + l];
    bool  v = dur[b * L_ + l] > 0.0f;      // padding <=> duration == 0
    float d = fabsf(tc - c);
    dist[j] = d;
    valid[j] = v;
    if (v) dmin = fminf(dmin, d);
  }
#pragma unroll
  for (int off = 16; off; off >>= 1) dmin = fminf(dmin, __shfl_xor(dmin, off));

  float w[16];
  float sum = 0.0f;
#pragma unroll
  for (int j = 0; j < 16; ++j) {
    float e = valid[j] ? __expf(-INV_TEMP * (dist[j] - dmin)) : 0.0f;
    w[j] = e;
    sum += e;
  }
#pragma unroll
  for (int off = 16; off; off >>= 1) sum += __shfl_xor(sum, off);
  float inv = 1.0f / sum;                  // sum >= 1 (dmin term contributes 1)

  _Float16* Wrow = W + ((size_t)b * Tpad + t) * L_;
#pragma unroll
  for (int j = 0; j < 16; ++j)
    Wrow[lane + 32 * j] = (_Float16)(w[j] * inv);
}

// ---------------- Kernel 4: batched WMMA GEMM  out[b] = W[b] @ F[b] ----------------
// Workgroup tile: 128(M frames) x 128(N feature dims); 8 waves as 4(M) x 2(N);
// each wave: 32x64 = 2x4 fragments of v_wmma_f32_16x16x32_f16.
// K loop (L=512, 16 steps of 32) fully unrolled: all loads become
// base + immediate IOFFSET (max 976B << 24-bit), zero per-step address VALU,
// and the scheduler software-pipelines loads across WMMA groups.
__global__ void __launch_bounds__(256)
lr_gemm(const _Float16* __restrict__ W, const _Float16* __restrict__ Ft,
        float* __restrict__ out, int T, int Tpad) {
  int lane = threadIdx.x & 31;
  int wave = threadIdx.x >> 5;
  int wm = wave & 3;      // 4 waves along M
  int wn = wave >> 2;     // 2 waves along N
  int b = blockIdx.z;
  int m_base = blockIdx.y * 128 + wm * 32;
  int n_base = blockIdx.x * 128 + wn * 64;

  const _Float16* Wb  = W  + (size_t)b * Tpad * L_;
  const _Float16* Ftb = Ft + (size_t)b * D_ * L_;
  float*          outb = out + (size_t)b * T * D_;

  int ln16 = lane & 15;
  int kh   = lane >> 4;   // 0 for lanes 0-15, 1 for lanes 16-31

  // Loop-invariant, lane-resolved base pointers.
  // A (16x32, contiguous K order shared with B): halves 0..7  <- K = k0 + kh*16 + 0..7
  //                         halves 8..15 <- K = k0 + kh*16 + 8 + 0..7
  const _Float16* aptr[2];
#pragma unroll
  for (int mf = 0; mf < 2; ++mf) {
    int m = m_base + mf * 16 + ln16;
    int mc = (m < T) ? m : (T - 1);        // clamp: padded M-tiles read row T-1
    aptr[mf] = Wb + (size_t)mc * L_ + kh * 16;
  }
  // B (32x16, same contiguous K order as A): lanes0-15 K=k0..k0+15, lanes16-31 K=k0+16..k0+31,
  // consecutive K in consecutive halves -> one contiguous v16h per fragment.
  const _Float16* bptr[4];
#pragma unroll
  for (int nf = 0; nf < 4; ++nf)
    bptr[nf] = Ftb + (size_t)(n_base + nf * 16 + ln16) * L_ + kh * 16;

  // One-time stream hint per row (global_prefetch_b8).
  __builtin_prefetch(aptr[0], 0, 1);
  __builtin_prefetch(aptr[1], 0, 1);
  __builtin_prefetch(bptr[0], 0, 1);
  __builtin_prefetch(bptr[2], 0, 1);

  v8f acc[2][4] = {};

#pragma unroll
  for (int k0 = 0; k0 < L_; k0 += 32) {
    v16h a[2];
#pragma unroll
    for (int mf = 0; mf < 2; ++mf) {
      v8h lo = *(const v8h*)(aptr[mf] + k0);
      v8h hi = *(const v8h*)(aptr[mf] + k0 + 8);
      a[mf] = __builtin_shufflevector(lo, hi, 0, 1, 2, 3, 4, 5, 6, 7,
                                      8, 9, 10, 11, 12, 13, 14, 15);
    }
    v16h bm[4];
#pragma unroll
    for (int nf = 0; nf < 4; ++nf)
      bm[nf] = *(const v16h*)(bptr[nf] + k0);

#pragma unroll
    for (int mf = 0; mf < 2; ++mf)
#pragma unroll
      for (int nf = 0; nf < 4; ++nf)
        acc[mf][nf] = __builtin_amdgcn_wmma_f32_16x16x32_f16(
            false, a[mf], false, bm[nf], (short)0, acc[mf][nf], false, false);
  }

  // f32 C/D layout: VGPR r -> M = r + 8*(lane>=16); N = lane&15
#pragma unroll
  for (int mf = 0; mf < 2; ++mf) {
    int mb = m_base + mf * 16 + kh * 8;
#pragma unroll
    for (int r = 0; r < 8; ++r) {
      int m = mb + r;
      if (m >= T) continue;
      float* orow = outb + (size_t)m * D_;
#pragma unroll
      for (int nf = 0; nf < 4; ++nf)
        orow[n_base + nf * 16 + ln16] = acc[mf][nf][r];
    }
  }
}

extern "C" void kernel_launch(void* const* d_in, const int* in_sizes, int n_in,
                              void* d_out, int out_size, void* d_ws, size_t ws_size,
                              hipStream_t stream) {
  const float* features  = (const float*)d_in[0];
  const float* durations = (const float*)d_in[1];
  // d_in[2] = padding_mask (bool) -- derived from durations==0 instead
  // d_in[3] = total_frames (device scalar) -- T recovered from out_size

  int T = out_size / (B_ * D_);
  if (T <= 0) return;
  int Mtiles = (T + 127) / 128;
  int Tpad = Mtiles * 128;

  // workspace layout (256B aligned sections)
  char* ws = (char*)d_ws;
  float* cum = (float*)ws;
  size_t off = (size_t)B_ * L_ * sizeof(float);
  off = (off + 255) & ~(size_t)255;
  _Float16* Ft = (_Float16*)(ws + off);
  off += (size_t)B_ * D_ * L_ * sizeof(_Float16);
  off = (off + 255) & ~(size_t)255;
  _Float16* W = (_Float16*)(ws + off);

  lr_cumsum<<<B_, L_, 0, stream>>>(durations, cum);
  lr_transpose<<<dim3(L_ / 32, D_ / 32, B_), dim3(32, 8), 0, stream>>>(features, Ft);
  lr_weights<<<dim3((T + 7) / 8, B_), 256, 0, stream>>>(cum, durations, W, T, Tpad);
  lr_gemm<<<dim3(D_ / 128, Mtiles, B_), 256, 0, stream>>>(W, Ft, (float*)d_out, T, Tpad);
}
